// SynthesisLayer_60035052863517
// MI455X (gfx1250) — matrix operand order repaired and checked by reading, not compiled, not run on hardware
//
#include <hip/hip_runtime.h>

typedef unsigned int u32;
typedef unsigned short u16;
typedef unsigned long long u64;
typedef __attribute__((ext_vector_type(4)))  u32    u32x4;
typedef __attribute__((ext_vector_type(16))) __bf16 v16bf;
typedef __attribute__((ext_vector_type(8)))  float  v8f;

struct __align__(16) A8 { u32x4 lo, hi; };

// pointer types for the async global->LDS builtin (signature learned from
// hipcc diagnostic: param0 = AS(1) int vector_size(16) *)
typedef int v4i __attribute__((vector_size(16)));
typedef __attribute__((address_space(1))) v4i* g_v4p;
typedef __attribute__((address_space(3))) v4i* l_v4p;

// ---- constants -------------------------------------------------------------
#define BATCH 16
#define CIN   512
#define COUT  512
#define HH    64
#define WDIM  512
#define AFFINE_WG 0.044194173824159216f /* 1/sqrt(512) */
#define GAIN  1.4142135623730951f
#define SLOPE 0.2f
#define CLAMP_V 256.0f

// workspace map (float/u32 units)
#define OFF_S      0          /* 16*512                    */
#define OFF_RS     8192       /* 1                         */
#define OFF_WNORM  8448       /* 512                       */
#define OFF_WSCALE 8960       /* 16*512                    */
#define OFF_WSQ9   17152      /* 512*512 -> ends 279296    */
#define OFF_XPAD   279552     /* 17,842,176 u32 (bf16 x2)  */
#define OFF_PACKA  18121728   /* 18,874,368 u32            */

static __device__ __forceinline__ u16 f2bf(float f) {
  u32 u = __builtin_bit_cast(u32, f);
  u32 r = (u + 0x7FFFu + ((u >> 16) & 1u)) >> 16;   // round-to-nearest-even
  return (u16)r;
}

// async 16B global->LDS copy (ASYNCcnt-tracked), per CDNA5 ISA
static __device__ __forceinline__ void async16(const void* g, void* l) {
#if __has_builtin(__builtin_amdgcn_global_load_async_to_lds_b128)
  __builtin_amdgcn_global_load_async_to_lds_b128(
      (g_v4p)(u64)g, (l_v4p)(u32)(u64)l, 0, 0);
#else
  asm volatile("global_load_async_to_lds_b128 %0, %1, off"
               :: "v"((u32)(u64)l), "v"(g) : "memory");
#endif
}

static __device__ __forceinline__ void wait_async0() {
#if __has_builtin(__builtin_amdgcn_s_wait_asynccnt)
  __builtin_amdgcn_s_wait_asynccnt(0);
#else
  asm volatile("s_wait_asynccnt 0x0" ::: "memory");
#endif
}

// ---- 1) affine: s[b,c] = dot(w[b,:], aw[c,:])*g + ab[c] --------------------
__global__ __launch_bounds__(256) void k_affine(const float* __restrict__ w,
                                                const float* __restrict__ aw,
                                                const float* __restrict__ ab,
                                                float* __restrict__ s) {
  int gid = blockIdx.x * 256 + threadIdx.x;          // 8192 threads
  int b = gid >> 9, c = gid & 511;
  const float* wr = w + b * WDIM;
  const float* ar = aw + c * WDIM;
  float acc = 0.f;
  for (int j = 0; j < WDIM; ++j) acc += wr[j] * ar[j];
  s[gid] = acc * AFFINE_WG + ab[c];
}

// ---- 2) rs = rsqrt(mean(s^2)) over all B*CIN -------------------------------
__global__ __launch_bounds__(256) void k_rs(const float* __restrict__ s,
                                            float* __restrict__ rs) {
  __shared__ float red[256];
  float acc = 0.f;
  for (int i = threadIdx.x; i < BATCH * CIN; i += 256) { float v = s[i]; acc += v * v; }
  red[threadIdx.x] = acc; __syncthreads();
  for (int st = 128; st > 0; st >>= 1) {
    if (threadIdx.x < st) red[threadIdx.x] += red[threadIdx.x + st];
    __syncthreads();
  }
  if (threadIdx.x == 0) rs[0] = rsqrtf(red[0] / (float)(BATCH * CIN));
}

// ---- 3) per-o: wsq9[o,c] = sum_taps cw^2 ; wnorm[o] = rsqrt(mean) ----------
__global__ __launch_bounds__(256) void k_wnorm(const float* __restrict__ cw,
                                               float* __restrict__ wsq9,
                                               float* __restrict__ wnorm) {
  __shared__ float red[256];
  int o = blockIdx.x;
  float local = 0.f;
  for (int c = threadIdx.x; c < CIN; c += 256) {
    const float* p = cw + (size_t)(o * CIN + c) * 9;
    float q = 0.f;
#pragma unroll
    for (int t = 0; t < 9; ++t) q += p[t] * p[t];
    wsq9[o * CIN + c] = q;
    local += q;
  }
  red[threadIdx.x] = local; __syncthreads();
  for (int st = 128; st > 0; st >>= 1) {
    if (threadIdx.x < st) red[threadIdx.x] += red[threadIdx.x + st];
    __syncthreads();
  }
  if (threadIdx.x == 0) wnorm[o] = rsqrtf(red[0] / (float)(CIN * 9));
}

// ---- 4) wscale[b,o] = wnorm[o]*dcoef[b,o]*input_gain -----------------------
__global__ __launch_bounds__(256) void k_dcoef(const float* __restrict__ s,
                                               const float* __restrict__ rs,
                                               const float* __restrict__ wsq9,
                                               const float* __restrict__ wnorm,
                                               const float* __restrict__ mag,
                                               float* __restrict__ wscale) {
  int gid = blockIdx.x * 256 + threadIdx.x;          // 8192 threads
  int b = gid >> 9, o = gid & 511;
  float r = rs[0];
  float ig = rsqrtf(mag[0]);
  const float* sr = s + b * CIN;
  const float* qr = wsq9 + o * CIN;
  float acc = 0.f;
  for (int c = 0; c < CIN; ++c) { float sm = sr[c] * r; acc += sm * sm * qr[c]; }
  float wn = wnorm[o];
  float d = rsqrtf(wn * wn * acc + 1e-8f);
  wscale[b * COUT + o] = wn * d * ig;
}

// ---- 5) pack modulated weights as bf16 WMMA A-fragments --------------------
// layout: u32 idx = ((((b*32+o16)*9+tap)*16+cc)*32+lane)*8 + v
// lane: M = lane&15, half = lane>>4 ; VGPR v holds K = 2v+16*half, K+1
__global__ __launch_bounds__(256) void k_pack(const float* __restrict__ cw,
                                              const float* __restrict__ s,
                                              const float* __restrict__ rs,
                                              const float* __restrict__ wscale,
                                              u32* __restrict__ packA) {
  u32 idx = blockIdx.x * 256 + threadIdx.x;          // 18,874,368 total
  u32 v = idx & 7u;      u32 rest = idx >> 3;
  u32 lane = rest & 31u; rest >>= 5;
  u32 cc = rest & 15u;   rest >>= 4;
  u32 tap = rest % 9u;   rest /= 9u;
  u32 o16 = rest & 31u;  u32 b = rest >> 5;
  int o = (int)(o16 * 16 + (lane & 15u));
  int hl = (int)(lane >> 4);
  int k0 = 2 * (int)v + 16 * hl;
  int c = (int)cc * 32 + k0;
  float r = rs[0];
  float sc = wscale[b * COUT + o];
  float f0 = cw[((size_t)o * CIN + c) * 9 + tap]     * (s[b * CIN + c] * r)     * sc;
  float f1 = cw[((size_t)o * CIN + c + 1) * 9 + tap] * (s[b * CIN + c + 1] * r) * sc;
  packA[idx] = ((u32)f2bf(f1) << 16) | (u32)f2bf(f0);
}

// ---- 6) pre-pack x as padded channel-tiled bf16: xpad[b][cc][66][66][32] ---
__global__ __launch_bounds__(256) void k_xpack(const float* __restrict__ x,
                                               u32* __restrict__ xpad) {
  u32 idx = blockIdx.x * 256 + threadIdx.x;          // 17,842,176 total
  u32 pr = idx & 15u;  u32 rest = idx >> 4;
  u32 w66 = rest % 66u; rest /= 66u;
  u32 h66 = rest % 66u; rest /= 66u;
  u32 cc = rest & 15u;  u32 b = rest >> 4;
  u32 val = 0u;
  if (h66 >= 1u && h66 <= 64u && w66 >= 1u && w66 <= 64u) {
    int c = (int)(cc * 32u + 2u * pr);
    const float* xp = x + (((size_t)b * CIN + c) * HH + (h66 - 1)) * HH + (w66 - 1);
    val = (u32)f2bf(xp[0]) | ((u32)f2bf(xp[HH * HH]) << 16);
  }
  xpad[idx] = val;
}

// ---- 7) zero the 66x66 pad border ------------------------------------------
__global__ __launch_bounds__(256) void k_border(float* __restrict__ out) {
  int idx = blockIdx.x * 256 + threadIdx.x;          // 16*512*260
  if (idx >= BATCH * COUT * 260) return;
  int p = idx % 260, bc = idx / 260;
  int h, w;
  if (p < 66)       { h = 0;           w = p;        }
  else if (p < 132) { h = 65;          w = p - 66;   }
  else if (p < 196) { h = p - 132 + 1; w = 0;        }
  else              { h = p - 196 + 1; w = 65;       }
  out[(size_t)bc * 66 * 66 + h * 66 + w] = 0.f;
}

// ---- 8) implicit-GEMM conv: async-LDS double buffer + bf16 WMMA ------------
// tile: M=64 cout x N=128 px (16w x 8h). 8 waves: m_sub = wave&3, n_half = wave>>2.
__global__ __launch_bounds__(256) void k_conv(const u16* __restrict__ xpad,
                                              const u32x4* __restrict__ packA,
                                              const float* __restrict__ bias,
                                              float* __restrict__ out) {
  // two 10x18x32 bf16 halo tiles, channel-contiguous per pixel
  __shared__ __align__(16) u16 xs[2 * 5760];         // 23040 B

  const int b = blockIdx.z, ob = blockIdx.y, sb = blockIdx.x;
  const int w0 = (sb & 3) * 16, h0 = (sb >> 2) * 8;
  const int tid = threadIdx.x, lane = tid & 31, wv = tid >> 5;
  const int m_sub = wv & 3, n_half = wv >> 2;
  const int o16 = ob * 4 + m_sub;
  const int hl = lane >> 4, nlo = lane & 15;

  // runtime (loop-invariant) part of every B-fragment LDS address; the
  // remaining tap/ns part is a compile-time immediate ds offset.
  const int bofs = (n_half * 72 + nlo) * 32 + hl * 16;   // u16 units

  // issue the async copy of c-chunk `cc` into LDS buffer `p`
  auto issueTile = [&](int cc, int p) {
    // global: contiguous 18px*64B row segments inside padded slab
    const char* gb = (const char*)xpad +
        (((size_t)(b * 16 + cc) * 4356) + (size_t)h0 * 66 + w0) * 64;
    u16* lb = &xs[p * 5760];
    for (int j = tid; j < 720; j += 256) {           // 720 x 16B chunks
      int row = j / 72, seg = j - row * 72;
      async16(gb + row * 4224 + seg * 16, lb + row * 576 + seg * 8);
    }
  };

  v8f acc[4] = {};

  issueTile(0, 0);
  for (int cc = 0; cc < 16; ++cc) {
    const int p = cc & 1;
    wait_async0();
    __syncthreads();                                 // buf p complete, buf p^1 free
    if (cc < 15) issueTile(cc + 1, p ^ 1);
    const u16* base = &xs[p * 5760 + bofs];

#pragma unroll
    for (int tap = 0; tap < 9; ++tap) {
      const int kh = tap / 3, kw = tap - kh * 3;
      // A fragment: 2 x b128 per lane, pre-swizzled in workspace
      size_t ia = (((((size_t)b * 32 + o16) * 9 + tap) * 16 + cc) * 32 + lane) * 2;
      __builtin_prefetch(&packA[ia + 64], 0, 1);
      A8 af; af.lo = packA[ia]; af.hi = packA[ia + 1];
      v16bf A = __builtin_bit_cast(v16bf, af);
      // preload all 4 B fragments (imm-offset ds loads), then 4 WMMAs
      v16bf Bv[4];
#pragma unroll
      for (int ns = 0; ns < 4; ++ns) {
        const u32x4* bp = (const u32x4*)(base + ((ns + kh) * 18 + kw) * 32);
        A8 bf; bf.lo = bp[0]; bf.hi = bp[1];
        Bv[ns] = __builtin_bit_cast(v16bf, bf);
      }
#pragma unroll
      for (int ns = 0; ns < 4; ++ns)
        acc[ns] = __builtin_amdgcn_wmma_f32_16x16x32_bf16(
            false, A, false, Bv[ns], (short)0, acc[ns], false, false);
    }
  }

  // epilogue: bias + lrelu*gain + clamp, store into 66x66 interior
  float bb[8];
#pragma unroll
  for (int r = 0; r < 8; ++r) bb[r] = bias[o16 * 16 + r + 8 * hl];
#pragma unroll
  for (int ns = 0; ns < 4; ++ns) {
    const int nrow = n_half * 4 + ns;
    const int h = h0 + nrow, w = w0 + nlo;
#pragma unroll
    for (int r = 0; r < 8; ++r) {
      const int o = o16 * 16 + r + 8 * hl;
      float t = acc[ns][r] + bb[r];
      t = (t >= 0.f ? t : t * SLOPE) * GAIN;
      t = fminf(fmaxf(t, -CLAMP_V), CLAMP_V);
      out[(((size_t)b * COUT + o) * 66 + (h + 1)) * 66 + (w + 1)] = t;
    }
  }
}

// ---- launcher ---------------------------------------------------------------
extern "C" void kernel_launch(void* const* d_in, const int* in_sizes, int n_in,
                              void* d_out, int out_size, void* d_ws, size_t ws_size,
                              hipStream_t stream) {
  const float* x    = (const float*)d_in[0];
  const float* w    = (const float*)d_in[1];
  const float* aw   = (const float*)d_in[2];
  const float* ab   = (const float*)d_in[3];
  const float* cw   = (const float*)d_in[4];
  const float* bias = (const float*)d_in[5];
  const float* mag  = (const float*)d_in[6];
  float* out = (float*)d_out;

  float* ws     = (float*)d_ws;
  float* s      = ws + OFF_S;
  float* rs     = ws + OFF_RS;
  float* wnorm  = ws + OFF_WNORM;
  float* wscale = ws + OFF_WSCALE;
  float* wsq9   = ws + OFF_WSQ9;
  u32*   xpad   = (u32*)(ws + OFF_XPAD);
  u32*   packA  = (u32*)(ws + OFF_PACKA);

  k_affine<<<32, 256, 0, stream>>>(w, aw, ab, s);
  k_rs    <<<1, 256, 0, stream>>>(s, rs);
  k_wnorm <<<COUT, 256, 0, stream>>>(cw, wsq9, wnorm);
  k_dcoef <<<32, 256, 0, stream>>>(s, rs, wsq9, wnorm, mag, wscale);
  k_pack  <<<73728, 256, 0, stream>>>(cw, s, rs, wscale, packA);
  k_xpack <<<69696, 256, 0, stream>>>(x, xpad);
  k_border<<<(BATCH * COUT * 260 + 255) / 256, 256, 0, stream>>>(out);
  k_conv  <<<dim3(32, 8, BATCH), 256, 0, stream>>>((const u16*)xpad,
                                                   (const u32x4*)packA, bias, out);
}
